// StandKvCache_2319282340612
// MI455X (gfx1250) — matrix operand order repaired and checked by reading, not compiled, not run on hardware
//
#include <hip/hip_runtime.h>
#include <math.h>

// ---------------------------------------------------------------------------
// Attention w/ KV cache for MI455X (gfx1250, wave32, WMMA bf16).
// All GEMM stages run on v_wmma_f32_16x16x32_bf16 with f32 accumulation.
// Tile staging uses CDNA5 async global->LDS copies (ASYNCcnt) + prefetch.
// ---------------------------------------------------------------------------

#define HIDDEN 2048
#define BATCH 4
#define Q_LEN 2048
#define PAST 2048
#define TKV (PAST + Q_LEN)  // 4096

typedef __attribute__((ext_vector_type(16))) __bf16        v16bf;
typedef __attribute__((ext_vector_type(8)))  float         v8f;
typedef __attribute__((ext_vector_type(4)))  unsigned int  u32x4;

union Frag { v16bf v; u32x4 u[2]; };

__device__ __forceinline__ unsigned short f2bf(float f) {
  unsigned int u = __float_as_uint(f);
  u += 0x7FFFu + ((u >> 16) & 1u);   // round-to-nearest-even
  return (unsigned short)(u >> 16);
}
__device__ __forceinline__ float bf2f(unsigned short h) {
  return __uint_as_float(((unsigned int)h) << 16);
}

// Async 16B global -> LDS copy (per-lane), tracked by ASYNCcnt.
__device__ __forceinline__ void async_copy_b128(unsigned lds_addr,
                                                const void* gptr) {
  unsigned long long ga = (unsigned long long)(uintptr_t)gptr;
  asm volatile("global_load_async_to_lds_b128 %0, %1, off"
               :: "v"(lds_addr), "v"(ga) : "memory");
}
__device__ __forceinline__ void wait_async0() {
  asm volatile("s_wait_asynccnt 0x0" ::: "memory");
}

// ---------------------------------------------------------------------------
// f32 -> bf16 pack
// ---------------------------------------------------------------------------
__global__ void cvt_f32_bf16_kernel(const float* __restrict__ src,
                                    unsigned short* __restrict__ dst, int n) {
  int i = blockIdx.x * blockDim.x + threadIdx.x;
  if (i < n) dst[i] = f2bf(src[i]);
}

// ---------------------------------------------------------------------------
// Copy past_k / past_v into f32 output cache region + bf16 compute buffer.
// transposed=1 writes dstB as [B, HIDDEN, TKV] (for the attn@v B-operand).
// ---------------------------------------------------------------------------
__global__ void copy_past_kernel(const float* __restrict__ src,
                                 float* __restrict__ dstF,
                                 unsigned short* __restrict__ dstB,
                                 int transposed) {
  long long i = (long long)blockIdx.x * blockDim.x + threadIdx.x; // < B*PAST*HIDDEN
  int d = (int)(i % HIDDEN);
  long long t = i / HIDDEN;
  int r = (int)(t % PAST);
  int b = (int)(t / PAST);
  float v = src[i];
  dstF[((size_t)b * TKV + r) * HIDDEN + d] = v;
  if (transposed)
    dstB[((size_t)b * HIDDEN + d) * TKV + r] = f2bf(v);
  else
    dstB[((size_t)b * TKV + r) * HIDDEN + d] = f2bf(v);
}

// ---------------------------------------------------------------------------
// Generic bf16 WMMA GEMM:  Y[row, n] = scale * sum_k A[row,k]*Wt[n,k] + bias[n]
//   A  : [Mrows, K] bf16 row-major (Mrows = BATCH * rows_per_batch)
//   Wt : [N, K]     bf16 row-major, per-batch base = Wt + b*wt_bstride
// Optional causal mask (scores stage): col > r + mask_off  ->  -inf
// Outputs (either may be null):
//   outF f32 at [(b*outF_T + outF_off + r)*N + n]
//   outB bf16, normal same layout, or transposed [(b*N + n)*outB_T + outB_off + r]
// Block tile 128x128, 8 waves (4 along M x 2 along N), wave tile 32x64,
// K staged through LDS in steps of 32 via async global->LDS copies.
// ---------------------------------------------------------------------------
#define BM 128
#define BN 128
#define BK 32

__global__ __launch_bounds__(256)
void gemm_bf16_kernel(const unsigned short* __restrict__ A,
                      const unsigned short* __restrict__ Wt,
                      size_t wt_bstride,
                      const float* __restrict__ bias,
                      float scale, int use_mask, int mask_off,
                      float* __restrict__ outF, int outF_T, int outF_off,
                      unsigned short* __restrict__ outB, int outB_T, int outB_off,
                      int outB_tr,
                      int N, int K, int rows_per_batch) {
  __shared__ alignas(16) unsigned short As[BM * BK];
  __shared__ alignas(16) unsigned short Ws[BN * BK];

  const int tid  = threadIdx.x;
  const int lane = tid & 31;
  const int wid  = tid >> 5;
  const int wm   = wid & 3;    // 4 waves along M
  const int wn   = wid >> 2;   // 2 waves along N
  const int row0 = blockIdx.y * BM;
  const int col0 = blockIdx.x * BN;
  const int half = lane >> 4;  // which 16-lane half
  const int l15  = lane & 15;

  const int b = row0 / rows_per_batch;  // BM divides rows_per_batch: no straddle
  const unsigned short* Wb = Wt + (size_t)b * wt_bstride;

  bool all_masked = false;
  if (use_mask) {
    int r_max = (row0 + BM - 1) % rows_per_batch;
    if (col0 > r_max + mask_off) all_masked = true;  // whole tile above diagonal
  }

  v8f acc[2][4] = {};
  if (!all_masked) {
    // Per-thread chunk ids (two 16B chunks of each tile per thread).
    const int c0r = tid >> 2,         c0o = (tid & 3) * 8;
    const int c1r = (tid + 256) >> 2, c1o = ((tid + 256) & 3) * 8;
    const unsigned lAs0 = (unsigned)(uintptr_t)&As[c0r * BK + c0o];
    const unsigned lWs0 = (unsigned)(uintptr_t)&Ws[c0r * BK + c0o];
    const unsigned lAs1 = (unsigned)(uintptr_t)&As[c1r * BK + c1o];
    const unsigned lWs1 = (unsigned)(uintptr_t)&Ws[c1r * BK + c1o];

    for (int k0 = 0; k0 < K; k0 += BK) {
      // Async-stage A/W tiles straight into LDS (no VGPR round trip).
      async_copy_b128(lAs0, &A[(size_t)(row0 + c0r) * K + k0 + c0o]);
      async_copy_b128(lWs0, &Wb[(size_t)(col0 + c0r) * K + k0 + c0o]);
      async_copy_b128(lAs1, &A[(size_t)(row0 + c1r) * K + k0 + c1o]);
      async_copy_b128(lWs1, &Wb[(size_t)(col0 + c1r) * K + k0 + c1o]);
      if (k0 + BK < K) {  // global_prefetch_b8 of next K tile into GL2
        __builtin_prefetch(&A[(size_t)(row0 + (tid & 127)) * K + k0 + BK], 0, 1);
        __builtin_prefetch(&Wb[(size_t)(col0 + (tid & 127)) * K + k0 + BK], 0, 1);
      }
      wait_async0();       // ASYNCcnt == 0: this wave's LDS writes landed
      __syncthreads();     // all waves' tiles visible

      // A fragment: lane l -> row l&15, 16B chunks at K = half*8 and 16+half*8
      Frag a[2], w[4];
      #pragma unroll
      for (int mi = 0; mi < 2; mi++) {
        const unsigned short* p = &As[(wm * 32 + mi * 16 + l15) * BK + half * 8];
        a[mi].u[0] = *(const u32x4*)p;
        a[mi].u[1] = *(const u32x4*)(p + 16);
      }
      // B fragment: lane l -> Wt row l&15, 16B chunks at K = half*8 and 16+half*8
      #pragma unroll
      for (int ni = 0; ni < 4; ni++) {
        const unsigned short* p = &Ws[(wn * 64 + ni * 16 + l15) * BK + half * 8];
        w[ni].u[0] = *(const u32x4*)p;
        w[ni].u[1] = *(const u32x4*)(p + 16);
      }

      #pragma unroll
      for (int mi = 0; mi < 2; mi++)
        #pragma unroll
        for (int ni = 0; ni < 4; ni++)
          acc[mi][ni] = __builtin_amdgcn_wmma_f32_16x16x32_bf16(
              false, a[mi].v, false, w[ni].v, (short)0, acc[mi][ni],
              false, false);
      __syncthreads();     // guard LDS reuse by next iteration's async stores
    }
  }

  // Epilogue. C/D layout: VGPR i, lane l -> (m = i + 8*half, n = l&15).
  #pragma unroll
  for (int mi = 0; mi < 2; mi++)
    #pragma unroll
    for (int ni = 0; ni < 4; ni++)
      #pragma unroll
      for (int i = 0; i < 8; i++) {
        int grow = row0 + wm * 32 + mi * 16 + half * 8 + i;
        int gcol = col0 + wn * 64 + ni * 16 + l15;
        int r = grow % rows_per_batch;
        float val;
        if (all_masked) {
          val = -INFINITY;
        } else {
          val = acc[mi][ni][i] * scale;
          if (bias) val += bias[gcol];
          if (use_mask && gcol > r + mask_off) val = -INFINITY;
        }
        if (outF)
          outF[((size_t)b * outF_T + outF_off + r) * N + gcol] = val;
        if (outB) {
          unsigned short hv = f2bf(val);
          if (outB_tr)
            outB[((size_t)b * N + gcol) * outB_T + outB_off + r] = hv;
          else
            outB[((size_t)b * outB_T + outB_off + r) * N + gcol] = hv;
        }
      }
}

// ---------------------------------------------------------------------------
// Row softmax over TKV bf16 scores, in place (masked entries are -inf -> 0).
// One 256-thread block per row; 16 columns per thread.
// ---------------------------------------------------------------------------
__global__ void softmax_kernel(unsigned short* __restrict__ S) {
  __shared__ float red[256];
  unsigned short* p = S + (size_t)blockIdx.x * TKV;
  const int tid = threadIdx.x;

  float vals[16];
  float lmax = -INFINITY;
  #pragma unroll
  for (int j = 0; j < 16; j++) {
    float v = bf2f(p[tid + j * 256]);
    vals[j] = v;
    lmax = fmaxf(lmax, v);
  }
  red[tid] = lmax;
  __syncthreads();
  for (int s = 128; s > 0; s >>= 1) {
    if (tid < s) red[tid] = fmaxf(red[tid], red[tid + s]);
    __syncthreads();
  }
  float m = red[0];
  __syncthreads();

  float lsum = 0.f;
  #pragma unroll
  for (int j = 0; j < 16; j++) {
    float e = __expf(vals[j] - m);
    vals[j] = e;
    lsum += e;
  }
  red[tid] = lsum;
  __syncthreads();
  for (int s = 128; s > 0; s >>= 1) {
    if (tid < s) red[tid] += red[tid + s];
    __syncthreads();
  }
  float inv = 1.0f / red[0];
  #pragma unroll
  for (int j = 0; j < 16; j++) p[tid + j * 256] = f2bf(vals[j] * inv);
}

// ---------------------------------------------------------------------------
extern "C" void kernel_launch(void* const* d_in, const int* in_sizes, int n_in,
                              void* d_out, int out_size, void* d_ws, size_t ws_size,
                              hipStream_t stream) {
  (void)in_sizes; (void)n_in; (void)out_size; (void)ws_size;

  const float* x  = (const float*)d_in[0];
  const float* pk = (const float*)d_in[1];
  const float* pv = (const float*)d_in[2];
  const float* Wq = (const float*)d_in[3];
  const float* bq = (const float*)d_in[4];
  const float* Wk = (const float*)d_in[5];
  const float* bk = (const float*)d_in[6];
  const float* Wv = (const float*)d_in[7];
  const float* bv = (const float*)d_in[8];
  const float* Wo = (const float*)d_in[9];
  const float* bo = (const float*)d_in[10];

  float* out  = (float*)d_out;                                 // [B, Tq, D]
  float* outK = out  + (size_t)BATCH * Q_LEN * HIDDEN;         // [B, TKV, D]
  float* outV = outK + (size_t)BATCH * TKV * HIDDEN;           // [B, TKV, D]

  // Workspace layout (bf16 = unsigned short elements)
  unsigned short* ws = (unsigned short*)d_ws;
  size_t o = 0;
  unsigned short* xb   = ws + o; o += (size_t)BATCH * Q_LEN * HIDDEN;
  unsigned short* wqb  = ws + o; o += (size_t)HIDDEN * HIDDEN;
  unsigned short* wkb  = ws + o; o += (size_t)HIDDEN * HIDDEN;
  unsigned short* wvb  = ws + o; o += (size_t)HIDDEN * HIDDEN;
  unsigned short* wob  = ws + o; o += (size_t)HIDDEN * HIDDEN;
  unsigned short* qb   = ws + o; o += (size_t)BATCH * Q_LEN * HIDDEN;
  unsigned short* kb   = ws + o; o += (size_t)BATCH * TKV * HIDDEN;   // [B,TKV,D]
  unsigned short* vbT  = ws + o; o += (size_t)BATCH * HIDDEN * TKV;   // [B,D,TKV]
  unsigned short* ctxb = ws + o; o += (size_t)BATCH * Q_LEN * HIDDEN;
  unsigned short* Sb   = ws + o; o += (size_t)BATCH * Q_LEN * TKV;    // [B,Tq,TKV]

  const int nX = BATCH * Q_LEN * HIDDEN;   // 16,777,216
  const int nW = HIDDEN * HIDDEN;          // 4,194,304

  cvt_f32_bf16_kernel<<<nX / 256, 256, 0, stream>>>(x,  xb,  nX);
  cvt_f32_bf16_kernel<<<nW / 256, 256, 0, stream>>>(Wq, wqb, nW);
  cvt_f32_bf16_kernel<<<nW / 256, 256, 0, stream>>>(Wk, wkb, nW);
  cvt_f32_bf16_kernel<<<nW / 256, 256, 0, stream>>>(Wv, wvb, nW);
  cvt_f32_bf16_kernel<<<nW / 256, 256, 0, stream>>>(Wo, wob, nW);

  const int nP = BATCH * PAST * HIDDEN;
  copy_past_kernel<<<nP / 256, 256, 0, stream>>>(pk, outK, kb, 0);
  copy_past_kernel<<<nP / 256, 256, 0, stream>>>(pv, outV, vbT, 1);

  const int Mrows = BATCH * Q_LEN;  // 8192
  dim3 blk(256);
  dim3 gProj(HIDDEN / BN, Mrows / BM);   // (16, 64)
  dim3 gScor(TKV / BN,    Mrows / BM);   // (32, 64)

  // q = x @ Wq^T + bq                       -> qb (bf16)
  gemm_bf16_kernel<<<gProj, blk, 0, stream>>>(
      xb, wqb, 0, bq, 1.0f, 0, 0,
      nullptr, 0, 0, qb, Q_LEN, 0, 0, HIDDEN, HIDDEN, Q_LEN);
  // k_new = x @ Wk^T + bk  -> f32 cache rows [PAST, TKV) and kb (bf16)
  gemm_bf16_kernel<<<gProj, blk, 0, stream>>>(
      xb, wkb, 0, bk, 1.0f, 0, 0,
      outK, TKV, PAST, kb, TKV, PAST, 0, HIDDEN, HIDDEN, Q_LEN);
  // v_new = x @ Wv^T + bv  -> f32 cache and vbT (bf16, transposed)
  gemm_bf16_kernel<<<gProj, blk, 0, stream>>>(
      xb, wvb, 0, bv, 1.0f, 0, 0,
      outV, TKV, PAST, vbT, TKV, PAST, 1, HIDDEN, HIDDEN, Q_LEN);

  // scores = (q @ k^T) * 1/sqrt(D), causal mask (col > r + PAST -> -inf)
  float scale = 1.0f / sqrtf((float)HIDDEN);
  gemm_bf16_kernel<<<gScor, blk, 0, stream>>>(
      qb, kb, (size_t)TKV * HIDDEN, nullptr, scale, 1, PAST,
      nullptr, 0, 0, Sb, Q_LEN, 0, 0, TKV, HIDDEN, Q_LEN);

  softmax_kernel<<<Mrows, 256, 0, stream>>>(Sb);

  // context = P @ v     (B operand = vbT rows, K = TKV)
  gemm_bf16_kernel<<<gProj, blk, 0, stream>>>(
      Sb, vbT, (size_t)HIDDEN * TKV, nullptr, 1.0f, 0, 0,
      nullptr, 0, 0, ctxb, Q_LEN, 0, 0, HIDDEN, TKV, Q_LEN);

  // output = context @ Wo^T + bo   -> f32 d_out
  gemm_bf16_kernel<<<gProj, blk, 0, stream>>>(
      ctxb, wob, 0, bo, 1.0f, 0, 0,
      out, Q_LEN, 0, nullptr, 0, 0, 0, HIDDEN, HIDDEN, Q_LEN);
}
